// BayesianBlock_84928683311850
// MI455X (gfx1250) — matrix operand order repaired and checked by reading, not compiled, not run on hardware
//
#include <hip/hip_runtime.h>
#include <hip/hip_bf16.h>
#include <math.h>

#define H 4096
#define MROWS 8192   // B*S = 4*2048

typedef __attribute__((ext_vector_type(16))) __bf16 v16bf;
typedef __attribute__((ext_vector_type(8)))  float  v8f;
typedef __attribute__((ext_vector_type(4)))  __bf16 bf16x4;

static __device__ __forceinline__ float softplus_f(float x) {
    // numerically stable log1p(exp(x))
    return fmaxf(x, 0.0f) + log1pf(expf(-fabsf(x)));
}

struct bf16pair { __bf16 hi, lo; };

static __device__ __forceinline__ bf16pair split_bf16(float w) {
    bf16pair p;
    p.hi = (__bf16)w;
    p.lo = (__bf16)(w - (float)p.hi);
    return p;
}

// ---------------------------------------------------------------------------
// Kernel 1: W = w_mu + softplus(w_rho) * eps_w, stored as bf16 hi/lo pair
// ---------------------------------------------------------------------------
__global__ __launch_bounds__(256) void prep_w_kernel(
    const float* __restrict__ mu, const float* __restrict__ rho,
    const float* __restrict__ eps,
    __bf16* __restrict__ whi, __bf16* __restrict__ wlo)
{
    const size_t i4 = (size_t)blockIdx.x * 256u + threadIdx.x;  // float4 index
    float4 m = ((const float4*)mu)[i4];
    float4 r = ((const float4*)rho)[i4];
    float4 e = ((const float4*)eps)[i4];
    bf16pair p0 = split_bf16(m.x + softplus_f(r.x) * e.x);
    bf16pair p1 = split_bf16(m.y + softplus_f(r.y) * e.y);
    bf16pair p2 = split_bf16(m.z + softplus_f(r.z) * e.z);
    bf16pair p3 = split_bf16(m.w + softplus_f(r.w) * e.w);
    bf16x4 hi, lo;
    hi[0] = p0.hi; lo[0] = p0.lo;
    hi[1] = p1.hi; lo[1] = p1.lo;
    hi[2] = p2.hi; lo[2] = p2.lo;
    hi[3] = p3.hi; lo[3] = p3.lo;
    ((bf16x4*)whi)[i4] = hi;
    ((bf16x4*)wlo)[i4] = lo;
}

// ---------------------------------------------------------------------------
// Kernel 2: b = b_mu + softplus(b_rho) * eps_b  (fp32, tiny)
// ---------------------------------------------------------------------------
__global__ __launch_bounds__(256) void prep_b_kernel(
    const float* __restrict__ mu, const float* __restrict__ rho,
    const float* __restrict__ eps, float* __restrict__ bias)
{
    const int i = blockIdx.x * 256 + threadIdx.x;
    if (i < H) bias[i] = mu[i] + softplus_f(rho[i]) * eps[i];
}

// ---------------------------------------------------------------------------
// Kernel 3: LayerNorm over last dim; emit h as bf16 hi/lo pair.
// One 256-thread block per row; each thread owns 4 float4 (16 elems).
// ---------------------------------------------------------------------------
__global__ __launch_bounds__(256) void ln_kernel(
    const float* __restrict__ x, const float* __restrict__ gamma,
    const float* __restrict__ beta,
    __bf16* __restrict__ hhi, __bf16* __restrict__ hlo)
{
    const int row = blockIdx.x;
    const int t = threadIdx.x;
    const float4* xr = (const float4*)(x + (size_t)row * H);

    float4 v[4];
    float s = 0.0f, ss = 0.0f;
#pragma unroll
    for (int i = 0; i < 4; ++i) {
        v[i] = xr[t + 256 * i];
        s  += v[i].x + v[i].y + v[i].z + v[i].w;
        ss += v[i].x * v[i].x + v[i].y * v[i].y + v[i].z * v[i].z + v[i].w * v[i].w;
    }

    __shared__ float2 red[256];
    red[t] = make_float2(s, ss);
    __syncthreads();
#pragma unroll
    for (int o = 128; o > 0; o >>= 1) {
        if (t < o) {
            red[t].x += red[t + o].x;
            red[t].y += red[t + o].y;
        }
        __syncthreads();
    }
    const float mean = red[0].x * (1.0f / H);
    const float var  = red[0].y * (1.0f / H) - mean * mean;
    const float rstd = rsqrtf(var + 1e-5f);

    bf16x4* ohi = (bf16x4*)(hhi + (size_t)row * H);
    bf16x4* olo = (bf16x4*)(hlo + (size_t)row * H);
#pragma unroll
    for (int i = 0; i < 4; ++i) {
        const int c4 = t + 256 * i;
        float4 g = ((const float4*)gamma)[c4];
        float4 b = ((const float4*)beta)[c4];
        bf16pair p0 = split_bf16((v[i].x - mean) * rstd * g.x + b.x);
        bf16pair p1 = split_bf16((v[i].y - mean) * rstd * g.y + b.y);
        bf16pair p2 = split_bf16((v[i].z - mean) * rstd * g.z + b.z);
        bf16pair p3 = split_bf16((v[i].w - mean) * rstd * g.w + b.w);
        bf16x4 hi, lo;
        hi[0] = p0.hi; lo[0] = p0.lo;
        hi[1] = p1.hi; lo[1] = p1.lo;
        hi[2] = p2.hi; lo[2] = p2.lo;
        hi[3] = p3.hi; lo[3] = p3.lo;
        ohi[c4] = hi;
        olo[c4] = lo;
    }
}

// ---------------------------------------------------------------------------
// Kernel 4: WMMA GEMM  y = h * W^T  (split-bf16, 3 products per tile)
//           + bias + exact-erf GELU + residual.
// Block = 128 threads = 4 waves (wave32); block tile 128x128, wave tile 64x64.
// ---------------------------------------------------------------------------
static __device__ __forceinline__ v16bf load_frag_a(const __bf16* p) {
    // 16-bit A-matrix layout (ISA 7.12.2): lane (M=lane%16) holds K chunks
    // [koff .. koff+7] and [koff+16 .. koff+23], koff = (lane/16)*8.
    // p points at row_base + k + koff.
    union { v16bf v; uint4 u[2]; } r;
    r.u[0] = *(const uint4*)(p);
    r.u[1] = *(const uint4*)(p + 16);
    return r.v;
}

static __device__ __forceinline__ v16bf load_frag_b(const __bf16* p) {
    // 16-bit B-matrix layout: lane (N=lane%16) holds K chunks
    // [koff .. koff+7] and [koff+16 .. koff+23], koff = (lane/16)*8.
    // p points at row_base + k + koff.
    union { v16bf v; uint4 u[2]; } r;
    r.u[0] = *(const uint4*)(p);
    r.u[1] = *(const uint4*)(p + 16);
    return r.v;
}

__global__ __launch_bounds__(128) void gemm_gelu_kernel(
    const __bf16* __restrict__ Ahi, const __bf16* __restrict__ Alo,
    const __bf16* __restrict__ Bhi, const __bf16* __restrict__ Blo,
    const float* __restrict__ bias, const float* __restrict__ resid,
    float* __restrict__ out)
{
    const int lane = threadIdx.x & 31;
    const int wave = threadIdx.x >> 5;
    const int m_base = blockIdx.y * 128 + (wave >> 1) * 64;
    const int n_base = blockIdx.x * 128 + (wave & 1) * 64;
    const int rlane = lane & 15;
    const int koff_a = (lane >> 4) << 3;   // A: 8-element interleave
    const int koff_b = (lane >> 4) << 3;   // B: 8-element interleave

    const v8f zacc = {0.f, 0.f, 0.f, 0.f, 0.f, 0.f, 0.f, 0.f};
    v8f acc[4][4];
#pragma unroll
    for (int mi = 0; mi < 4; ++mi)
#pragma unroll
        for (int ni = 0; ni < 4; ++ni) acc[mi][ni] = zacc;

    size_t aoff[4], boff[4];
#pragma unroll
    for (int mi = 0; mi < 4; ++mi)
        aoff[mi] = (size_t)(m_base + mi * 16 + rlane) * H + koff_a;
#pragma unroll
    for (int ni = 0; ni < 4; ++ni)
        boff[ni] = (size_t)(n_base + ni * 16 + rlane) * H + koff_b;

#pragma unroll 1
    for (int k = 0; k < H; k += 32) {
        v16bf ah[4], al[4], bh[4], bl[4];
#pragma unroll
        for (int mi = 0; mi < 4; ++mi) {
            ah[mi] = load_frag_a(Ahi + aoff[mi] + k);
            al[mi] = load_frag_a(Alo + aoff[mi] + k);
        }
#pragma unroll
        for (int ni = 0; ni < 4; ++ni) {
            bh[ni] = load_frag_b(Bhi + boff[ni] + k);
            bl[ni] = load_frag_b(Blo + boff[ni] + k);
        }
#pragma unroll
        for (int mi = 0; mi < 4; ++mi) {
#pragma unroll
            for (int ni = 0; ni < 4; ++ni) {
                acc[mi][ni] = __builtin_amdgcn_wmma_f32_16x16x32_bf16(
                    false, ah[mi], false, bh[ni], (short)0, acc[mi][ni], false, false);
                acc[mi][ni] = __builtin_amdgcn_wmma_f32_16x16x32_bf16(
                    false, ah[mi], false, bl[ni], (short)0, acc[mi][ni], false, false);
                acc[mi][ni] = __builtin_amdgcn_wmma_f32_16x16x32_bf16(
                    false, al[mi], false, bh[ni], (short)0, acc[mi][ni], false, false);
            }
        }
    }

    // Epilogue: C/D layout -> lane = N (lane&15), VGPR r = row (lane>>4)*8 + r
    const float kInvSqrt2 = 0.70710678118654752f;
#pragma unroll
    for (int mi = 0; mi < 4; ++mi) {
#pragma unroll
        for (int ni = 0; ni < 4; ++ni) {
            const int n = n_base + ni * 16 + (lane & 15);
            const int m0 = m_base + mi * 16 + (lane >> 4) * 8;
            const float bv = bias[n];
#pragma unroll
            for (int r = 0; r < 8; ++r) {
                const size_t idx = (size_t)(m0 + r) * H + n;
                float v = acc[mi][ni][r] + bv;
                float g = 0.5f * v * (1.0f + erff(v * kInvSqrt2));
                out[idx] = resid[idx] + g;
            }
        }
    }
}

// ---------------------------------------------------------------------------
// Host launcher
// ---------------------------------------------------------------------------
extern "C" void kernel_launch(void* const* d_in, const int* in_sizes, int n_in,
                              void* d_out, int out_size, void* d_ws, size_t ws_size,
                              hipStream_t stream) {
    (void)in_sizes; (void)n_in; (void)out_size; (void)ws_size;
    const float* x      = (const float*)d_in[0];
    const float* gamma  = (const float*)d_in[1];
    const float* beta   = (const float*)d_in[2];
    const float* w_mu   = (const float*)d_in[3];
    const float* w_rho  = (const float*)d_in[4];
    const float* b_mu   = (const float*)d_in[5];
    const float* b_rho  = (const float*)d_in[6];
    const float* eps_w  = (const float*)d_in[7];
    const float* eps_b  = (const float*)d_in[8];
    float* out = (float*)d_out;

    // workspace layout
    char* ws = (char*)d_ws;
    const size_t szW = (size_t)H * H * sizeof(__bf16);      // 32 MiB
    const size_t szH = (size_t)MROWS * H * sizeof(__bf16);  // 64 MiB
    __bf16* Whi = (__bf16*)(ws);
    __bf16* Wlo = (__bf16*)(ws + szW);
    __bf16* Hhi = (__bf16*)(ws + 2 * szW);
    __bf16* Hlo = (__bf16*)(ws + 2 * szW + szH);
    float*  bia = (float*)(ws + 2 * szW + 2 * szH);

    // 1) materialize W as bf16 hi/lo     (H*H/4 float4 elems / 256 thr)
    prep_w_kernel<<<(H * (size_t)H) / (4 * 256), 256, 0, stream>>>(w_mu, w_rho, eps_w, Whi, Wlo);
    // 2) bias
    prep_b_kernel<<<(H + 255) / 256, 256, 0, stream>>>(b_mu, b_rho, eps_b, bia);
    // 3) LayerNorm -> bf16 hi/lo
    ln_kernel<<<MROWS, 256, 0, stream>>>(x, gamma, beta, Hhi, Hlo);
    // 4) WMMA GEMM + bias + GELU + residual
    dim3 grid(H / 128, MROWS / 128);
    gemm_gelu_kernel<<<grid, 128, 0, stream>>>(Hhi, Hlo, Whi, Wlo, bia, x, out);
}
